// Attention_90941637525668
// MI455X (gfx1250) — matrix operand order repaired and checked by reading, not compiled, not run on hardware
//
#include <hip/hip_runtime.h>
#include <hip/hip_bf16.h>

// ---- problem constants ----
#define DDIM   1024
#define TDIM   2048
#define BDIM   32
#define NROWS  (BDIM * TDIM)     // 65536 rows of the fused (b,t) GEMM
#define MBLK   64                // rows per block
#define LDSS   1040              // A-panel LDS row stride in bf16 (1024 + 32B pad)
#define BSTRIDE 520              // B-chunk LDS row stride in bf16 (512 + 16B TDM pad)
#define ABYTES (MBLK * LDSS * 2)     // 133120
#define BBYTES (MBLK * BSTRIDE * 2)  // 66560 per K-half buffer

typedef __bf16 bf16_t;
typedef bf16_t v4bf   __attribute__((ext_vector_type(4)));
typedef bf16_t v8bf   __attribute__((ext_vector_type(8)));
typedef bf16_t v16bf  __attribute__((ext_vector_type(16)));
typedef float  v8f    __attribute__((ext_vector_type(8)));
typedef unsigned int u32;
typedef u32    v4u    __attribute__((ext_vector_type(4)));
typedef int    v4i    __attribute__((ext_vector_type(4)));
typedef int    v8i    __attribute__((ext_vector_type(8)));

// -------------------------------------------------------------------------
// Kernel 0: convert W_e = W[:, D:2D] (fp32, row stride 2048) -> bf16 [1024 x 1024]
// -------------------------------------------------------------------------
__global__ __launch_bounds__(256) void we_to_bf16(const float* __restrict__ W,
                                                  bf16_t* __restrict__ Wbf) {
    int idx = blockIdx.x * 256 + threadIdx.x;      // 0 .. 262143, 4 elems each
    int d  = idx >> 8;
    int kq = (idx & 255) << 2;
    float4 wv = *(const float4*)(W + (size_t)d * 2048 + 1024 + kq);
    v4bf o;
    o[0] = (bf16_t)wv.x; o[1] = (bf16_t)wv.y; o[2] = (bf16_t)wv.z; o[3] = (bf16_t)wv.w;
    *(v4bf*)(Wbf + ((size_t)d << 10) + kq) = o;
}

// -------------------------------------------------------------------------
// Kernel 1: hproj[b,d] = dot(h[b,:], W[d,0:1024]) + bias[d]   (fp32, tiny)
// -------------------------------------------------------------------------
__global__ __launch_bounds__(256) void hproj_kernel(const float* __restrict__ hidden,
                                                    const float* __restrict__ W,
                                                    const float* __restrict__ bias,
                                                    float* __restrict__ hproj) {
    int gw   = (blockIdx.x * 256 + threadIdx.x) >> 5;   // 0 .. 32767
    int lane = threadIdx.x & 31;
    int b = gw >> 10;
    int d = gw & 1023;
    const float* wrow = W + (size_t)d * 2048;
    const float* hrow = hidden + (size_t)b * 1024;
    float acc = 0.f;
#pragma unroll
    for (int i = 0; i < 8; ++i) {
        int k = (lane + i * 32) * 4;
        float4 wv = *(const float4*)(wrow + k);
        float4 hv = *(const float4*)(hrow + k);
        acc += wv.x * hv.x + wv.y * hv.y + wv.z * hv.z + wv.w * hv.w;
    }
#pragma unroll
    for (int off = 16; off > 0; off >>= 1) acc += __shfl_xor(acc, off, 32);
    if (lane == 0) hproj[gw] = acc + bias[d];
}

// -------------------------------------------------------------------------
// TDM helper: DMA a 64-row x 512-col bf16 tile (memory row stride 1024 elems)
// into LDS; pad 16 B after every 1024 B -> LDS row stride 520 bf16.
// -------------------------------------------------------------------------
__device__ __forceinline__ void tdm_load_b_chunk(const bf16_t* gsrc, u32 ldsOff) {
    unsigned long long ga = (unsigned long long)(size_t)gsrc;
    v4u g0;
    g0[0] = 1u;                                            // count=1, user mode
    g0[1] = ldsOff;                                        // lds_addr (bytes)
    g0[2] = (u32)ga;                                       // global_addr[31:0]
    g0[3] = (u32)((ga >> 32) & 0x01FFFFFFu) | 0x80000000u; // global_addr[56:32] | type=2

    v8i g1;
    g1[0] = (1 << 16)   // data_size = 1 -> 2 bytes
          | (1 << 20)   // pad_enable
          | (7 << 22)   // pad_interval: every 256 DWORDs (1024 B = one 512-elem row)
          | (3 << 25);  // pad_amount: 4 DWORDs (16 B)
    g1[1] = (int)(512u << 16);   // tensor_dim0[15:0] in bits[63:48]
    g1[2] = (int)(64u  << 16);   // tensor_dim1[15:0] in bits[95:80]
    g1[3] = (int)(512u << 16);   // tile_dim0 in bits[127:112]
    g1[4] = 64;                  // tile_dim1 (tile_dim2 = 0)
    g1[5] = 1024;                // tensor_dim0_stride[31:0] (elems)
    g1[6] = 0;
    g1[7] = 0;

#if __has_include(<hip/amd_detail/amd_gfx1250_TDM.h>)
    v4i g2 = {0, 0, 0, 0};
    v4i g3 = {0, 0, 0, 0};
    v8i g4 = {0, 0, 0, 0, 0, 0, 0, 0};
    __builtin_amdgcn_tensor_load_to_lds(g0, g1, g2, g3, g4, 0);   // 6-arg toolchain
#else
    v4i g2 = {0, 0, 0, 0};
    v4i g3 = {0, 0, 0, 0};
    __builtin_amdgcn_tensor_load_to_lds(g0, g1, g2, g3, 0);       // 5-arg toolchain (ROCm 7.2)
#endif
}

// -------------------------------------------------------------------------
// Kernel 2: fused  scores[row] = sum_d relu( enc[row,:]@We[d,:] + hproj[b,d] ) * v[d]
// Block: 64 rows, full N (16 groups of 64 cols), full K=1024 via WMMA bf16.
// B staged by the Tensor Data Mover in K-half chunks, double-buffered so the
// DMA of chunk i+1 overlaps the WMMA stream consuming chunk i (TENSORcnt<=1).
// -------------------------------------------------------------------------
__global__ __launch_bounds__(256) void attn_energy_scores(
    const float*  __restrict__ enc,     // [65536,1024] fp32
    const bf16_t* __restrict__ Wbf,     // [1024,1024] bf16 (W_e)
    const float*  __restrict__ hproj,   // [32,1024]  (includes bias)
    const float*  __restrict__ v,       // [1024]
    float*        __restrict__ scores)  // [65536], pre-zeroed
{
    extern __shared__ char smem[];
    bf16_t* As  = (bf16_t*)smem;                     // [64][LDSS]
    bf16_t* Bs0 = (bf16_t*)(smem + ABYTES);          // [64][BSTRIDE]  K-half 0
    bf16_t* Bs1 = (bf16_t*)(smem + ABYTES + BBYTES); // [64][BSTRIDE]  K-half 1

    const int tid  = threadIdx.x;
    const int lane = tid & 31;
    const int w    = tid >> 5;       // wave 0..7
    const int mt   = w & 3;          // m-tile (16 rows each)
    const int nh   = w >> 2;         // n-half (32 cols of the 64-col group)
    const int row0 = blockIdx.x * MBLK;
    const int b    = row0 >> 11;     // row0 / TDIM (MBLK divides TDIM -> uniform b)

    const u32 bs0Off = (u32)(size_t)(void*)Bs0;

    // prefetch chunk 0 (group 0, K-half 0) -> buf0; DMA overlaps A staging below
    if (w == 0) tdm_load_b_chunk(Wbf, bs0Off);

    // ---- stage A panel: enc rows row0..row0+63, all K, fp32 -> bf16 ----
#pragma unroll 4
    for (int i = 0; i < 64; ++i) {
        int c  = tid + 256 * i;           // chunk of 4 floats; 16384 chunks
        int r  = c >> 8;
        int kq = (c & 255) << 2;
        float4 ev = *(const float4*)(enc + (size_t)(row0 + r) * 1024 + kq);
        v4bf o;
        o[0] = (bf16_t)ev.x; o[1] = (bf16_t)ev.y; o[2] = (bf16_t)ev.z; o[3] = (bf16_t)ev.w;
        *(v4bf*)(As + r * LDSS + kq) = o;
    }

    // per-lane fragment base pointers (contiguous k = 16*half + s per lane, same for A and B)
    const int col  = lane & 15;
    const int half = lane >> 4;
    const bf16_t* aBase   = As + (mt * 16 + col) * LDSS + half * 16;
    const bf16_t* b0Base0 = Bs0 + (nh * 32 + col) * BSTRIDE + half * 16;
    const bf16_t* b1Base0 = b0Base0 + 16 * BSTRIDE;
    const bf16_t* b0Base1 = Bs1 + (nh * 32 + col) * BSTRIDE + half * 16;
    const bf16_t* b1Base1 = b0Base1 + 16 * BSTRIDE;

    float s[8] = {0.f, 0.f, 0.f, 0.f, 0.f, 0.f, 0.f, 0.f};

    for (int ng = 0; ng < 16; ++ng) {
        v8f c0 = {0.f, 0.f, 0.f, 0.f, 0.f, 0.f, 0.f, 0.f};
        v8f c1 = {0.f, 0.f, 0.f, 0.f, 0.f, 0.f, 0.f, 0.f};

#pragma unroll
        for (int h = 0; h < 2; ++h) {
            __syncthreads();   // prior reads of the buffer we are about to refill are done
            if (w == 0) {
                const int ci = ng * 2 + h;         // current chunk (already in flight)
                if (ci < 31) {
                    const int nci = ci + 1;        // prefetch next chunk into other buffer
                    tdm_load_b_chunk(Wbf + (size_t)(nci >> 1) * 65536 + (size_t)(nci & 1) * 512,
                                     bs0Off + (u32)(nci & 1) * BBYTES);
                    __builtin_amdgcn_s_wait_tensorcnt(1);   // chunk ci retired, nci in flight
                } else {
                    __builtin_amdgcn_s_wait_tensorcnt(0);   // last chunk
                }
            }
            __syncthreads();   // chunk ci visible to all waves

            const bf16_t* pb0 = h ? b0Base1 : b0Base0;
            const bf16_t* pb1 = h ? b1Base1 : b1Base0;
            const bf16_t* pa  = aBase + h * 512;

#pragma unroll 4
            for (int ks = 0; ks < 16; ++ks) {
                const int kc = ks * 32;
                // A frag: 16 consecutive K at kc (+16 for upper lane half)
                v8bf alo = *(const v8bf*)(pa + kc);
                v8bf ahi = *(const v8bf*)(pa + kc + 8);
                v16bf a = __builtin_shufflevector(alo, ahi,
                    0,1,2,3,4,5,6,7,8,9,10,11,12,13,14,15);
                // B frags: 16 consecutive K at kc (+16 for upper lane half)
                v8bf p0 = *(const v8bf*)(pb0 + kc);
                v8bf p1 = *(const v8bf*)(pb0 + kc + 8);
                v16bf bb0 = __builtin_shufflevector(p0, p1,
                    0,1,2,3,4,5,6,7,8,9,10,11,12,13,14,15);
                v8bf q0 = *(const v8bf*)(pb1 + kc);
                v8bf q1 = *(const v8bf*)(pb1 + kc + 8);
                v16bf bb1 = __builtin_shufflevector(q0, q1,
                    0,1,2,3,4,5,6,7,8,9,10,11,12,13,14,15);

                c0 = __builtin_amdgcn_wmma_f32_16x16x32_bf16(false, a, false, bb0,
                                                             (short)0, c0, false, false);
                c1 = __builtin_amdgcn_wmma_f32_16x16x32_bf16(false, a, false, bb1,
                                                             (short)0, c1, false, false);
            }
        }

        // ---- fused epilogue: + hproj, relu, dot with v ----
        const int nbase = ng * 64 + nh * 32;
        float hv0 = hproj[b * 1024 + nbase + col];
        float hv1 = hproj[b * 1024 + nbase + 16 + col];
        float vv0 = v[nbase + col];
        float vv1 = v[nbase + 16 + col];
#pragma unroll
        for (int r = 0; r < 8; ++r) {
            float e0 = c0[r] + hv0; e0 = e0 > 0.f ? e0 : 0.f;
            float e1 = c1[r] + hv1; e1 = e1 > 0.f ? e1 : 0.f;
            s[r] += e0 * vv0 + e1 * vv1;
        }
    }

    // reduce 16 lanes (one column set) -> per-row partial, combine across n-halves
#pragma unroll
    for (int r = 0; r < 8; ++r) {
        float t = s[r];
        t += __shfl_xor(t, 1, 32);
        t += __shfl_xor(t, 2, 32);
        t += __shfl_xor(t, 4, 32);
        t += __shfl_xor(t, 8, 32);
        if (col == 0) {
            int row = row0 + mt * 16 + r + 8 * half;  // C layout: VGPR r = M r / r+8
            atomicAdd(scores + row, t);
        }
    }
}

// -------------------------------------------------------------------------
// Kernel 3: softmax over T per batch row; out[b,0,t]
// -------------------------------------------------------------------------
__global__ __launch_bounds__(256) void softmax_t(const float* __restrict__ scores,
                                                 float* __restrict__ out) {
    __shared__ float red[256];
    const float* srow = scores + (size_t)blockIdx.x * TDIM;
    float vals[8];
    float lmax = -1e30f;
#pragma unroll
    for (int i = 0; i < 8; ++i) {
        vals[i] = srow[threadIdx.x + 256 * i];
        lmax = fmaxf(lmax, vals[i]);
    }
    red[threadIdx.x] = lmax; __syncthreads();
    for (int st = 128; st > 0; st >>= 1) {
        if (threadIdx.x < st) red[threadIdx.x] = fmaxf(red[threadIdx.x], red[threadIdx.x + st]);
        __syncthreads();
    }
    float m = red[0]; __syncthreads();
    float lsum = 0.f;
#pragma unroll
    for (int i = 0; i < 8; ++i) { vals[i] = __expf(vals[i] - m); lsum += vals[i]; }
    red[threadIdx.x] = lsum; __syncthreads();
    for (int st = 128; st > 0; st >>= 1) {
        if (threadIdx.x < st) red[threadIdx.x] += red[threadIdx.x + st];
        __syncthreads();
    }
    float inv = 1.f / red[0];
#pragma unroll
    for (int i = 0; i < 8; ++i)
        out[(size_t)blockIdx.x * TDIM + threadIdx.x + 256 * i] = vals[i] * inv;
}

// -------------------------------------------------------------------------
extern "C" void kernel_launch(void* const* d_in, const int* in_sizes, int n_in,
                              void* d_out, int out_size, void* d_ws, size_t ws_size,
                              hipStream_t stream) {
    const float* hidden = (const float*)d_in[0];   // [32,1024]
    const float* enc    = (const float*)d_in[1];   // [32,2048,1024]
    const float* W      = (const float*)d_in[2];   // [1024,2048]
    const float* bias   = (const float*)d_in[3];   // [1024]
    const float* v      = (const float*)d_in[4];   // [1024]
    float* out          = (float*)d_out;           // [32,1,2048]

    char*   ws     = (char*)d_ws;                  // needs ~2.4 MB
    bf16_t* Wbf    = (bf16_t*)ws;                              // 2 MB
    float*  hproj  = (float*)(ws + (2u << 20));                // 128 KB
    float*  scores = (float*)(ws + (2u << 20) + (128u << 10)); // 256 KB

    we_to_bf16<<<1024, 256, 0, stream>>>(W, Wbf);
    hproj_kernel<<<4096, 256, 0, stream>>>(hidden, W, bias, hproj);
    hipMemsetAsync(scores, 0, NROWS * sizeof(float), stream);

    size_t lds_bytes = (size_t)ABYTES + 2 * (size_t)BBYTES;    // 266,240 B (< 320 KB WGP LDS)
    attn_energy_scores<<<NROWS / MBLK, 256, lds_bytes, stream>>>(enc, Wbf, hproj, v, scores);

    softmax_t<<<BDIM, 256, 0, stream>>>(scores, out);
}
